// Conv_hg_90022514524500
// MI455X (gfx1250) — compile-verified, matrix operand order checked by reading, not run on hardware
//
#include <hip/hip_runtime.h>
#include <hip/hip_bf16.h>

// ---------------------------------------------------------------------------
// Problem constants (match reference)
// ---------------------------------------------------------------------------
#define DMODEL 256
#define N_EV_C   50000
#define N_OBJ_C 100000
#define N_NODES_C 150000
#define E1_C 400000LL
#define E2_C 800000LL

#define APAD 40   // ushorts per LDS A row (32 data + 8 pad) -> 80B, 16B aligned
#define WPAD 40   // ushorts per LDS W row

typedef __attribute__((ext_vector_type(16))) __bf16 v16bf;
typedef __attribute__((ext_vector_type(8)))  float  v8f;
typedef __attribute__((ext_vector_type(8)))  unsigned v8u;

// float -> bf16 (round to nearest even), raw ushort
__device__ __forceinline__ unsigned short f2bf(float f) {
    unsigned u = __float_as_uint(f);
    u += 0x7FFFu + ((u >> 16) & 1u);
    return (unsigned short)(u >> 16);
}

__device__ __forceinline__ float lrelu(float x) { return x > 0.f ? x : 0.2f * x; }

__device__ __forceinline__ void atomicMaxF(float* addr, float v) {
    if (v >= 0.f) atomicMax((int*)addr, __float_as_int(v));
    else          atomicMin((unsigned*)addr, __float_as_uint(v));
}

// CDNA5 async global->LDS copy (ASYNCcnt tracked), 16 bytes per lane
__device__ __forceinline__ void async_ld_b128(unsigned lds_off, unsigned long long gaddr) {
    asm volatile("global_load_async_to_lds_b128 %0, %1, off"
                 :: "v"(lds_off), "v"(gaddr) : "memory");
}
__device__ __forceinline__ void wait_async0() {
    asm volatile("s_wait_asynccnt 0" ::: "memory");
}

// ---------------------------------------------------------------------------
// W pre-pass: Wt[n][k] = bf16(W[k][n])  (transposed, bf16), 256x256
// ---------------------------------------------------------------------------
__global__ __launch_bounds__(256)
void convW_kernel(const float* __restrict__ W, unsigned short* __restrict__ Wt) {
    int i = blockIdx.x * 256 + threadIdx.x;        // 0..65535
    int n = i >> 8, k = i & 255;
    Wt[(n << 8) | k] = f2bf(W[(k << 8) | n]);
}

// ---------------------------------------------------------------------------
// Fused GEMM: C[N,256] = epilogue( A[N,256] @ W[256,256] )
//   mode 0: identity, 1: LN(lrelu(acc+bias)), 2: LN(lrelu(acc+bias)+res)
// bf16 WMMA 16x16x32, f32 accum. Block=256thr(8 waves), 32-row stripe.
// W slabs streamed to LDS with async b128 copies, double buffered.
// ---------------------------------------------------------------------------
__global__ __launch_bounds__(256)
void gemm_fused(const float* __restrict__ A, const unsigned short* __restrict__ Wbf,
                const float* __restrict__ bias, const float* __restrict__ gamma,
                const float* __restrict__ beta, const float* __restrict__ res,
                float* __restrict__ C, int Nrows, int mode) {
    __shared__ __align__(16) unsigned short As[2][32][APAD];
    __shared__ __align__(16) unsigned short Ws[2][DMODEL][WPAD];
    __shared__ float Cst[32][DMODEL + 1];

    const int tid  = threadIdx.x;
    const int wv   = tid >> 5;
    const int lane = tid & 31;
    const int hf   = lane >> 4;
    const int ml   = lane & 15;
    const int rowBase = blockIdx.x * 32;

    // stage one 32x32 A slab (fp32 -> packed bf16), 1 float4 per thread
    auto stageA = [&](int bufI, int k0) {
        int r = tid >> 3, ch = tid & 7;
        int gr = rowBase + r;
        uint2 packed;
        if (gr < Nrows) {
            const float4 f = *(const float4*)(A + (size_t)gr * DMODEL + k0 + ch * 4);
            packed.x = ((unsigned)f2bf(f.y) << 16) | (unsigned)f2bf(f.x);
            packed.y = ((unsigned)f2bf(f.w) << 16) | (unsigned)f2bf(f.z);
        } else {
            packed.x = 0u; packed.y = 0u;
        }
        *(uint2*)&As[bufI][r][ch * 4] = packed;
    };

    // issue async copy of one 32k x 256n W slab (bf16, already transposed)
    auto issueW = [&](int bufI, int k0) {
        unsigned ldsBase = (unsigned)(uintptr_t)&Ws[bufI][tid][0];
        unsigned long long g =
            (unsigned long long)(uintptr_t)(Wbf + ((size_t)tid << 8) + k0);
        #pragma unroll
        for (int c = 0; c < 4; ++c)
            async_ld_b128(ldsBase + c * 16, g + (unsigned long long)(c * 16));
    };

    v8f acc00 = {0,0,0,0,0,0,0,0}, acc01 = acc00, acc10 = acc00, acc11 = acc00;

    // prologue
    stageA(0, 0);
    issueW(0, 0);
    wait_async0();
    __syncthreads();

    int cur = 0;
    const int n0 = wv * 32 + ml;
    const int n1 = n0 + 16;
    for (int s = 0; s < 8; ++s) {
        if (s < 7) {                    // prefetch next slab into other buffer
            issueW(cur ^ 1, (s + 1) * 32);
            stageA(cur ^ 1, (s + 1) * 32);
        }
        // fragments per ISA 16-bit 16x32 layout:
        // VGPR v<4: K=2v,2v+1 (+8 for lanes 16-31); v>=4: K=16+2(v-4)... (+8)
        v8u ua0, ua1, ub0, ub1;
        #pragma unroll
        for (int v = 0; v < 8; ++v) {
            int kk = ((v < 4) ? (2 * v) : (16 + 2 * (v - 4))) + hf * 8;
            ua0[v] = *(const unsigned*)&As[cur][ml][kk];
            ua1[v] = *(const unsigned*)&As[cur][16 + ml][kk];
            ub0[v] = *(const unsigned*)&Ws[cur][n0][kk];
            ub1[v] = *(const unsigned*)&Ws[cur][n1][kk];
        }
        v16bf fa0 = __builtin_bit_cast(v16bf, ua0);
        v16bf fa1 = __builtin_bit_cast(v16bf, ua1);
        v16bf fb0 = __builtin_bit_cast(v16bf, ub0);
        v16bf fb1 = __builtin_bit_cast(v16bf, ub1);

        acc00 = __builtin_amdgcn_wmma_f32_16x16x32_bf16(false, fa0, false, fb0, (short)0, acc00, false, false);
        acc01 = __builtin_amdgcn_wmma_f32_16x16x32_bf16(false, fa0, false, fb1, (short)0, acc01, false, false);
        acc10 = __builtin_amdgcn_wmma_f32_16x16x32_bf16(false, fa1, false, fb0, (short)0, acc10, false, false);
        acc11 = __builtin_amdgcn_wmma_f32_16x16x32_bf16(false, fa1, false, fb1, (short)0, acc11, false, false);

        wait_async0();                  // own async W copies for next slab done
        __syncthreads();                // everyone done reading cur + writes visible
        cur ^= 1;
    }

    // C/D layout: VGPR r -> row = hf*8 + r, col = ml (per 16x16 tile)
    #pragma unroll
    for (int r = 0; r < 8; ++r) {
        int rowh = hf * 8 + r;
        Cst[rowh][wv * 32 + ml]           = acc00[r];
        Cst[rowh][wv * 32 + ml + 16]      = acc01[r];
        Cst[16 + rowh][wv * 32 + ml]      = acc10[r];
        Cst[16 + rowh][wv * 32 + ml + 16] = acc11[r];
    }
    __syncthreads();

    if (mode == 0) {
        for (int i = tid; i < 32 * DMODEL; i += 256) {
            int r = i >> 8, c = i & 255;
            int gr = rowBase + r;
            if (gr < Nrows) C[(size_t)gr * DMODEL + c] = Cst[r][c];
        }
        return;
    }

    // LayerNorm epilogue: 8 lanes per row
    const int r   = tid >> 3;
    const int sub = tid & 7;
    const int gr  = rowBase + r;
    float sum = 0.f, sumsq = 0.f;
    float vals[32];
    #pragma unroll
    for (int j = 0; j < 32; ++j) {
        int c = sub + 8 * j;
        float t = lrelu(Cst[r][c] + bias[c]);
        if (mode == 2 && gr < Nrows) t += res[(size_t)gr * DMODEL + c];
        vals[j] = t;
        sum += t; sumsq += t * t;
    }
    #pragma unroll
    for (int off = 1; off < 8; off <<= 1) {
        sum   += __shfl_xor(sum,   off, 8);
        sumsq += __shfl_xor(sumsq, off, 8);
    }
    float mean = sum * (1.f / 256.f);
    float var  = sumsq * (1.f / 256.f) - mean * mean;
    float inv  = rsqrtf(var + 1e-5f);
    if (gr < Nrows) {
        #pragma unroll
        for (int j = 0; j < 32; ++j) {
            int c = sub + 8 * j;
            C[(size_t)gr * DMODEL + c] = (vals[j] - mean) * inv * gamma[c] + beta[c];
        }
    }
}

// ---------------------------------------------------------------------------
// Memory-bound kernels (HBM-roofline bound; unchanged)
// ---------------------------------------------------------------------------
__global__ void fill_kernel(float* __restrict__ p, float v, long long n) {
    long long i = (long long)blockIdx.x * blockDim.x + threadIdx.x;
    if (i < n) p[i] = v;
}

__global__ __launch_bounds__(256)
void rowdot_kernel(const float* __restrict__ M, const float* __restrict__ a,
                   float* __restrict__ s, int N) {
    int wv = threadIdx.x >> 5, lane = threadIdx.x & 31;
    int row = blockIdx.x * 8 + wv;
    if (row >= N) return;
    float acc = 0.f;
    #pragma unroll
    for (int c = lane; c < DMODEL; c += 32)
        acc += M[(size_t)row * DMODEL + c] * a[c];
    #pragma unroll
    for (int off = 16; off; off >>= 1) acc += __shfl_xor(acc, off, 32);
    if (lane == 0) s[row] = acc;
}

__global__ __launch_bounds__(256)
void scatter_rows4(const float* __restrict__ src, const int* __restrict__ srcIdx,
                   const int* __restrict__ dstIdx, float* __restrict__ dst, long long E) {
    long long t = (long long)blockIdx.x * blockDim.x + threadIdx.x;
    if (t >= E * 64) return;
    long long e = t >> 6;
    int c4 = (int)(t & 63) * 4;
    const float* sp = src + (size_t)srcIdx[e] * DMODEL + c4;
    float* dp = dst + (size_t)dstIdx[e] * DMODEL + c4;
    atomicAdd(dp + 0, sp[0]); atomicAdd(dp + 1, sp[1]);
    atomicAdd(dp + 2, sp[2]); atomicAdd(dp + 3, sp[3]);
}

__global__ __launch_bounds__(256)
void scatter_out4(const float* __restrict__ ef, const float* __restrict__ attn,
                  const int* __restrict__ node, const int* __restrict__ edge,
                  float* __restrict__ out, long long E) {
    long long t = (long long)blockIdx.x * blockDim.x + threadIdx.x;
    if (t >= E * 64) return;
    long long e = t >> 6;
    int c4 = (int)(t & 63) * 4;
    float a = attn[e];
    const float* sp = ef + (size_t)edge[e] * DMODEL + c4;
    float* dp = out + (size_t)node[e] * DMODEL + c4;
    atomicAdd(dp + 0, a * sp[0]); atomicAdd(dp + 1, a * sp[1]);
    atomicAdd(dp + 2, a * sp[2]); atomicAdd(dp + 3, a * sp[3]);
}

__global__ void count_kernel(const int* __restrict__ edge, float* __restrict__ cnt, long long E) {
    long long e = (long long)blockIdx.x * blockDim.x + threadIdx.x;
    if (e < E) atomicAdd(&cnt[edge[e]], 1.f);
}

__global__ void divcnt_kernel(float* __restrict__ ef, const float* __restrict__ cnt, long long n) {
    long long i = (long long)blockIdx.x * blockDim.x + threadIdx.x;
    if (i < n) ef[i] /= fmaxf(cnt[i >> 8], 1.f);
}

__global__ void score_kernel(const float* __restrict__ sn, const float* __restrict__ se,
                             const int* __restrict__ node, const int* __restrict__ edge,
                             float* __restrict__ score, float* __restrict__ smax, long long E) {
    long long e = (long long)blockIdx.x * blockDim.x + threadIdx.x;
    if (e >= E) return;
    int nd = node[e];
    float sc = lrelu(sn[nd] + se[edge[e]]);
    score[e] = sc;
    atomicMaxF(&smax[nd], sc);
}

__global__ void expz_kernel(const float* __restrict__ score, const float* __restrict__ smax,
                            const int* __restrict__ node, float* __restrict__ ebuf,
                            float* __restrict__ z, long long E) {
    long long e = (long long)blockIdx.x * blockDim.x + threadIdx.x;
    if (e >= E) return;
    int nd = node[e];
    float ex = __expf(score[e] - smax[nd]);
    ebuf[e] = ex;
    atomicAdd(&z[nd], ex);
}

__global__ void attn_kernel(float* __restrict__ ebuf, const float* __restrict__ z,
                            const int* __restrict__ node, float* __restrict__ attn_copy, long long E) {
    long long e = (long long)blockIdx.x * blockDim.x + threadIdx.x;
    if (e >= E) return;
    float a = ebuf[e] / fmaxf(z[node[e]], 1e-9f);
    ebuf[e] = a;
    if (attn_copy) attn_copy[e] = a;
}

__global__ void lrelu_kernel(const float* __restrict__ in, float* __restrict__ out, long long n) {
    long long i = (long long)blockIdx.x * blockDim.x + threadIdx.x;
    if (i < n) out[i] = lrelu(in[i]);
}

// ---------------------------------------------------------------------------
// Host orchestration
// ---------------------------------------------------------------------------
static inline int nblk(long long n, int b = 256) { return (int)((n + b - 1) / b); }

extern "C" void kernel_launch(void* const* d_in, const int* in_sizes, int n_in,
                              void* d_out, int out_size, void* d_ws, size_t ws_size,
                              hipStream_t stream) {
    (void)in_sizes; (void)n_in; (void)out_size; (void)ws_size;
    const int NEV = N_EV_C, NOBJ = N_OBJ_C, NN = N_NODES_C;
    const long long E1 = E1_C, E2 = E2_C;

    const float* object_X = (const float*)d_in[0];
    const float* event_X  = (const float*)d_in[1];
    const float* Wo = (const float*)d_in[2];  const float* bo  = (const float*)d_in[3];
    const float* go = (const float*)d_in[4];  const float* bon = (const float*)d_in[5];
    const float* We = (const float*)d_in[6];  const float* be  = (const float*)d_in[7];
    const float* ge = (const float*)d_in[8];  const float* ben = (const float*)d_in[9];
    const float* Wu = (const float*)d_in[10]; const float* bu  = (const float*)d_in[11];
    const float* Wl = (const float*)d_in[12]; const float* bl  = (const float*)d_in[13];
    const float* g1 = (const float*)d_in[14]; const float* b1  = (const float*)d_in[15];
    const float* g2 = (const float*)d_in[16]; const float* b2  = (const float*)d_in[17];
    const float* Wh1 = (const float*)d_in[18]; const float* ah1 = (const float*)d_in[19];
    const float* Wh2 = (const float*)d_in[20]; const float* ah2 = (const float*)d_in[21];
    const int* oe_ev   = (const int*)d_in[22];
    const int* oe_obj  = (const int*)d_in[23];
    const int* hg_node = (const int*)d_in[24];
    const int* hg_edge = (const int*)d_in[25];

    // workspace layout (floats)
    const size_t ND = (size_t)NN * DMODEL;
    float* ws = (float*)d_ws;
    float* X   = ws;                               // [NN, D] concat(ev, obj2)
    float* M   = X + ND;                           // [NN, D] XW / attention out
    float* H   = M + ND;                           // [NN, D] msg, then layer-1 h
    float* B1  = H + ND;                           // [NOBJ, D]
    float* EF  = B1 + (size_t)NOBJ * DMODEL;       // [NEV, D]
    float* cnt    = EF + (size_t)NEV * DMODEL;     // [NEV]
    float* s_node = cnt + NEV;                     // [NN]
    float* s_edge = s_node + NN;                   // [NEV]
    float* score  = s_edge + NEV;                  // [E2]
    float* ebuf   = score + E2;                    // [E2]
    float* zbuf   = ebuf + E2;                     // [NN]
    float* smax   = zbuf + NN;                     // [NN]
    unsigned short* Wbf = (unsigned short*)(smax + NN);   // [256*256] bf16, 16B-aligned

    float* out_h    = (float*)d_out;               // [NN, D]
    float* out_attn = out_h + ND;                  // [E2]
    float* X_obj = X + (size_t)NEV * DMODEL;

    auto gemm = [&](const float* A, const float* W, const float* bias,
                    const float* g, const float* b, const float* res,
                    float* C, int N, int mode) {
        convW_kernel<<<256, 256, 0, stream>>>(W, Wbf);
        gemm_fused<<<(N + 31) / 32, 256, 0, stream>>>(A, Wbf, bias, g, b, res, C, N, mode);
    };

    // --- projection chains ---
    gemm(event_X, We, be, ge, ben, nullptr, X, NEV, 1);        // ev -> X[0:NEV]
    gemm(object_X, Wo, bo, go, bon, nullptr, B1, NOBJ, 1);     // obj -> B1
    fill_kernel<<<nblk((long long)NOBJ * DMODEL), 256, 0, stream>>>(H, 0.f, (long long)NOBJ * DMODEL);
    scatter_rows4<<<nblk(E1 * 64), 256, 0, stream>>>(X, oe_ev, oe_obj, H, E1);
    gemm(H, Wu, bu, g1, b1, B1, X_obj, NOBJ, 2);               // obj1 -> X_obj
    gemm(X_obj, Wl, bl, g2, b2, X_obj, B1, NOBJ, 2);           // obj2 -> B1
    hipMemcpyAsync(X_obj, B1, (size_t)NOBJ * DMODEL * sizeof(float),
                   hipMemcpyDeviceToDevice, stream);

    // hyperedge incidence counts (shared by both layers)
    fill_kernel<<<nblk(NEV), 256, 0, stream>>>(cnt, 0.f, NEV);
    count_kernel<<<nblk(E2), 256, 0, stream>>>(hg_edge, cnt, E2);

    auto hgnn = [&](const float* Xin, const float* W, const float* a,
                    float* XW, float* OUT, float* hout, float* attn_copy) {
        gemm(Xin, W, nullptr, nullptr, nullptr, nullptr, XW, NN, 0);
        fill_kernel<<<nblk((long long)NEV * DMODEL), 256, 0, stream>>>(EF, 0.f, (long long)NEV * DMODEL);
        scatter_rows4<<<nblk(E2 * 64), 256, 0, stream>>>(XW, hg_node, hg_edge, EF, E2);
        divcnt_kernel<<<nblk((long long)NEV * DMODEL), 256, 0, stream>>>(EF, cnt, (long long)NEV * DMODEL);
        rowdot_kernel<<<(NN + 7) / 8, 256, 0, stream>>>(XW, a, s_node, NN);
        rowdot_kernel<<<(NEV + 7) / 8, 256, 0, stream>>>(EF, a + DMODEL, s_edge, NEV);
        fill_kernel<<<nblk(NN), 256, 0, stream>>>(smax, -3.402823466e38f, NN);
        score_kernel<<<nblk(E2), 256, 0, stream>>>(s_node, s_edge, hg_node, hg_edge, score, smax, E2);
        fill_kernel<<<nblk(NN), 256, 0, stream>>>(zbuf, 0.f, NN);
        expz_kernel<<<nblk(E2), 256, 0, stream>>>(score, smax, hg_node, ebuf, zbuf, E2);
        attn_kernel<<<nblk(E2), 256, 0, stream>>>(ebuf, zbuf, hg_node, attn_copy, E2);
        fill_kernel<<<nblk((long long)NN * DMODEL), 256, 0, stream>>>(OUT, 0.f, (long long)NN * DMODEL);
        scatter_out4<<<nblk(E2 * 64), 256, 0, stream>>>(EF, ebuf, hg_node, hg_edge, OUT, E2);
        lrelu_kernel<<<nblk((long long)NN * DMODEL), 256, 0, stream>>>(OUT, hout, (long long)NN * DMODEL);
    };

    // layer 1: X -> H   (XW and OUT reuse M)
    hgnn(X, Wh1, ah1, M, M, H, nullptr);
    // layer 2: H -> d_out (XW and OUT reuse X; layer-2 attn -> d_out tail)
    hgnn(H, Wh2, ah2, X, X, out_h, out_attn);
}
